// LSTM_75127567942059
// MI455X (gfx1250) — matrix operand order repaired and checked by reading, not compiled, not run on hardware
//
#include <hip/hip_runtime.h>
#include <hip/hip_bf16.h>

// Problem constants (from reference)
#define BATCH 128
#define SEQT  1024
#define DIN   256
#define HIDD  512
#define ODIM  256
#define KDIM  (DIN + HIDD)   // 768 fused K for weight rows [x-part | h-part]
#define NWG   32             // persistent workgroups
#define UPW   (HIDD / NWG)   // 16 hidden units per WG
#define NGCOL (4 * UPW)      // 64 gate-columns per WG (i,g,f,o x 16 units)

typedef __bf16 bf16_t;
typedef __attribute__((ext_vector_type(16))) __bf16 v16bf;
typedef __attribute__((ext_vector_type(8)))  __bf16 v8bf;
typedef __attribute__((ext_vector_type(8)))  float  v8f;
typedef __attribute__((ext_vector_type(4)))  unsigned int v4u;
typedef __attribute__((ext_vector_type(8)))  int v8i;
typedef __attribute__((ext_vector_type(4)))  int v4i;

// ---------------------------------------------------------------------------
// WMMA operand loaders; A and B share one contiguous k order (not the ISA table order).
// A: 16x32 bf16 from row-major [M][K]:
//   lane L: m = mbase + (L&15); elem j -> k = kbase + (L>>4)*16 + j (contiguous)
// B: 32x16 bf16 from row-major [N][K]:
//   lane L: n = nbase + (L&15); elem j -> k = kbase + (L>>4)*16 + j (contiguous)
// ---------------------------------------------------------------------------
__device__ __forceinline__ v16bf load_a_tile(const bf16_t* base, int ld, int mbase, int kbase) {
  const int lane = threadIdx.x & 31;
  const bf16_t* p = base + (mbase + (lane & 15)) * ld + kbase + ((lane >> 4) << 4);
  return *(const v16bf*)p;
}

__device__ __forceinline__ v16bf load_b_tile(const bf16_t* base, int ld, int nbase, int kbase) {
  const int lane = threadIdx.x & 31;
  const bf16_t* p = base + (nbase + (lane & 15)) * ld + kbase + ((lane >> 4) << 4);
  return *(const v16bf*)p;   // 32 contiguous bytes
}

__device__ __forceinline__ v8f wmma_bf16(v16bf a, v16bf b, v8f c) {
  return __builtin_amdgcn_wmma_f32_16x16x32_bf16(false, a, false, b, (short)0, c, false, false);
}

__device__ __forceinline__ float fsigmoid(float x) { return 1.0f / (1.0f + __expf(-x)); }

// ---------------------------------------------------------------------------
// Tensor Data Mover: 1-D contiguous copy global -> LDS, n_units of 8 bytes.
// D# per CDNA5 ISA Sec 8.3-8.5:
//   group0: [1:0]=count=1, [63:32]=lds_addr, [120:64]=global_addr, [127:126]=type=2
//   group1: [17:16]=data_size=3 (8B), [79:48]=tensor_dim0=n, [111:80]=tensor_dim1=1,
//           [127:112]=tile_dim0=n, [143:128]=tile_dim1=1, [207:160]=dim0_stride=n
// Issued wave-uniform; completion tracked with TENSORcnt.
// ---------------------------------------------------------------------------
__device__ __forceinline__ void tdm_load_1d_to_lds(unsigned lds_addr, const void* gsrc,
                                                   unsigned n_units) {
  unsigned long long ga = (unsigned long long)gsrc;
  v4u g0;
  g0[0] = 1u;                                   // count=1, user mode, no gather
  g0[1] = lds_addr;                             // D#.lds_addr (bytes)
  g0[2] = (unsigned)ga;                         // global_addr[31:0]
  g0[3] = (unsigned)(ga >> 32) | (2u << 30);    // global_addr[56:32] | type=2
  v8i g1;
  g1[0] = (int)(3u << 16);                      // workgroup_mask=0, data_size=8B
  g1[1] = (int)((n_units & 0xFFFFu) << 16);     // tensor_dim0[15:0]
  g1[2] = (int)((n_units >> 16) | (1u << 16));  // tensor_dim0[31:16] | tensor_dim1=1
  g1[3] = (int)((n_units & 0xFFFFu) << 16);     // tensor_dim1[31:16]=0 | tile_dim0=n
  g1[4] = 1;                                    // tile_dim1=1, tile_dim2=0
  g1[5] = (int)n_units;                         // tensor_dim0_stride[31:0]
  g1[6] = 0;                                    // stride0[47:32] | dim1_stride[15:0]
  g1[7] = 0;                                    // dim1_stride[47:16]
  v4i z = {0, 0, 0, 0};
#if defined(__clang_major__) && (__clang_major__ >= 23)
  v8i z8 = {0, 0, 0, 0, 0, 0, 0, 0};
  __builtin_amdgcn_tensor_load_to_lds(g0, g1, z, z, z8, 0);   // 6-arg toolchain
#else
  __builtin_amdgcn_tensor_load_to_lds(g0, g1, z, z, 0);       // 5-arg (ROCm 7.2)
#endif
}

// ---------------------------------------------------------------------------
// Pack params: fused bf16 weight [4*H][768] (rows gate-major; cols already
// [x-part | h-part] since W = [H, D+H] row-major), f32 fused bias, bf16 Wout.
// Also zero-initializes the grid barrier and both h double-buffers so every
// launch (and every graph replay) is deterministic.
// ---------------------------------------------------------------------------
__global__ void pack_params(const float* __restrict__ Wi, const float* __restrict__ Wg,
                            const float* __restrict__ Wf, const float* __restrict__ Wo,
                            const float* __restrict__ bi, const float* __restrict__ bg,
                            const float* __restrict__ bfv, const float* __restrict__ bo,
                            const float* __restrict__ Wout,
                            bf16_t* __restrict__ Wcomb, float* __restrict__ bcomb,
                            bf16_t* __restrict__ Woutbf, bf16_t* __restrict__ hbuf,
                            unsigned* __restrict__ bar) {
  const int stride = gridDim.x * blockDim.x;
  const int t0 = blockIdx.x * blockDim.x + threadIdx.x;
  const int WSZ = HIDD * KDIM;
  for (int i = t0; i < 4 * WSZ; i += stride) {
    const float* src = (i < WSZ) ? Wi : (i < 2 * WSZ) ? Wg : (i < 3 * WSZ) ? Wf : Wo;
    Wcomb[i] = (bf16_t)src[i % WSZ];
  }
  for (int i = t0; i < 4 * HIDD; i += stride) {
    const float* src = (i < HIDD) ? bi : (i < 2 * HIDD) ? bg : (i < 3 * HIDD) ? bfv : bo;
    bcomb[i] = src[i % HIDD];
  }
  for (int i = t0; i < ODIM * HIDD; i += stride) Woutbf[i] = (bf16_t)Wout[i];
  for (int i = t0; i < 2 * BATCH * HIDD; i += stride) hbuf[i] = (bf16_t)0.0f;  // h0 = 0
  for (int i = t0; i < 64; i += stride) bar[i] = 0u;
}

// x [B][T][D] f32  ->  xbf [T][B][D] bf16  (contiguous 64KB tile per step)
__global__ void pack_x(const float* __restrict__ x, bf16_t* __restrict__ xbf) {
  const long long total = (long long)SEQT * BATCH * DIN;
  const long long stride = (long long)gridDim.x * blockDim.x;
  for (long long i = blockIdx.x * (long long)blockDim.x + threadIdx.x; i < total; i += stride) {
    int d = (int)(i % DIN);
    long long r = i / DIN;
    int b = (int)(r % BATCH);
    int t = (int)(r / BATCH);
    xbf[i] = (bf16_t)x[((long long)b * SEQT + t) * DIN + d];
  }
}

// ---------------------------------------------------------------------------
// Persistent LSTM scan. Grid = 32 WGs x 256 threads (8 wave32s).
// WG w owns hidden units [16w, 16w+16) across all four gates.
//   LDS: sX [128][256] + sH [128][512] activations (192KB), sW [64][768] (96KB).
// Per step: wave0 issues TDM tensor_load_to_lds for x_t and h_t, waits
// TENSORcnt; each wave computes a 16x64 gate tile (24 K x 4 WMMA); gate
// nonlinearities + cell update per-lane in f32 registers; h_{t+1} slice goes
// to the other global h buffer; one grid barrier per step.
// ---------------------------------------------------------------------------
__global__ void __launch_bounds__(256, 1)
lstm_scan(const bf16_t* __restrict__ xbf, const bf16_t* __restrict__ Wcomb,
          const float* __restrict__ bcomb, bf16_t* __restrict__ hbuf,
          unsigned* __restrict__ bar) {
  __shared__ __align__(64) bf16_t sX[BATCH * DIN];    //  65536 B
  __shared__ __align__(64) bf16_t sH[BATCH * HIDD];   // 131072 B
  __shared__ __align__(64) bf16_t sW[NGCOL * KDIM];   //  98304 B  (total 288KB <= 320KB/WGP)

  const int tid  = threadIdx.x;
  const int wave = tid >> 5;
  const int lane = tid & 31;
  const int wg   = blockIdx.x;
  const int u0   = wg * UPW;
  const int mbase = wave * 16;

  const unsigned ldsX = (unsigned)(unsigned long long)(const void*)sX;
  const unsigned ldsH = (unsigned)(unsigned long long)(const void*)sH;
  const unsigned ldsW = (unsigned)(unsigned long long)(const void*)sW;

  // Weight slice: 4 contiguous blocks (16 consecutive Wcomb rows per gate) via TDM.
  if (wave == 0) {
#pragma unroll
    for (int g = 0; g < 4; ++g) {
      tdm_load_1d_to_lds(ldsW + (unsigned)(g * UPW * KDIM * 2),
                         Wcomb + (size_t)(g * HIDD + u0) * KDIM,
                         (UPW * KDIM * 2) / 8);                 // 3072 x 8B
    }
  }

  // Per-lane gate biases (constant across steps; n = lane&15 maps to unit u)
  const int ul = lane & 15;
  const int u  = u0 + ul;
  const float b_i = bcomb[0 * HIDD + u];
  const float b_g = bcomb[1 * HIDD + u];
  const float b_f = bcomb[2 * HIDD + u];
  const float b_o = bcomb[3 * HIDD + u];

  v8f cstate = {0, 0, 0, 0, 0, 0, 0, 0};   // cell state lives in registers (C/D layout)

  for (int t = 0; t < SEQT; ++t) {
    const bf16_t* hread  = hbuf + (size_t)(t & 1) * BATCH * HIDD;
    bf16_t*       hwrite = hbuf + (size_t)((t + 1) & 1) * BATCH * HIDD;

    // ---- stage x_t and h_t into LDS with the Tensor Data Mover ----
    if (wave == 0) {
      tdm_load_1d_to_lds(ldsX, xbf + (size_t)t * BATCH * DIN, (BATCH * DIN * 2) / 8);
      tdm_load_1d_to_lds(ldsH, hread, (BATCH * HIDD * 2) / 8);
    }
    // Prefetch next step's x tile toward L2 while the TDM works (global_prefetch_b8)
    if (t + 1 < SEQT) {
      const char* nx = (const char*)(xbf + (size_t)(t + 1) * BATCH * DIN);
      __builtin_prefetch(nx + tid * 256, 0, 1);
    }
    if (wave == 0) __builtin_amdgcn_s_wait_tensorcnt(0);
    __syncthreads();

    // ---- GEMM: wave's 16-row M-strip x 64 gate-cols, K = 256 (x) + 512 (h) ----
    v8f acc0 = {0, 0, 0, 0, 0, 0, 0, 0};
    v8f acc1 = {0, 0, 0, 0, 0, 0, 0, 0};
    v8f acc2 = {0, 0, 0, 0, 0, 0, 0, 0};
    v8f acc3 = {0, 0, 0, 0, 0, 0, 0, 0};
    for (int k = 0; k < DIN; k += 32) {
      v16bf av = load_a_tile(sX, DIN, mbase, k);
      acc0 = wmma_bf16(av, load_b_tile(sW, KDIM,  0, k), acc0);  // i
      acc1 = wmma_bf16(av, load_b_tile(sW, KDIM, 16, k), acc1);  // g
      acc2 = wmma_bf16(av, load_b_tile(sW, KDIM, 32, k), acc2);  // f
      acc3 = wmma_bf16(av, load_b_tile(sW, KDIM, 48, k), acc3);  // o
    }
    for (int k = 0; k < HIDD; k += 32) {
      v16bf av = load_a_tile(sH, HIDD, mbase, k);
      acc0 = wmma_bf16(av, load_b_tile(sW, KDIM,  0, DIN + k), acc0);
      acc1 = wmma_bf16(av, load_b_tile(sW, KDIM, 16, DIN + k), acc1);
      acc2 = wmma_bf16(av, load_b_tile(sW, KDIM, 32, DIN + k), acc2);
      acc3 = wmma_bf16(av, load_b_tile(sW, KDIM, 48, DIN + k), acc3);
    }

    // ---- gates + cell update (pure per-lane VALU; all 4 gates at same slot) ----
#pragma unroll
    for (int v = 0; v < 8; ++v) {
      float ig = fsigmoid(acc0[v] + b_i);
      float gg = tanhf(acc1[v] + b_g);
      float fg = fsigmoid(acc2[v] + b_f);
      float og = fsigmoid(acc3[v] + b_o);
      float c  = fg * cstate[v] + ig * gg;
      cstate[v] = c;
      float h  = og * tanhf(c);
      int m = mbase + ((lane >> 4) << 3) + v;   // C/D layout row
      hwrite[(size_t)m * HIDD + u] = (bf16_t)h;
    }

    // ---- grid barrier (sense-reversal; one per step; h is double-buffered) ----
    __threadfence();
    __syncthreads();
    if (tid == 0) {
      unsigned target = (unsigned)(t + 1) * NWG;
      unsigned prev = __hip_atomic_fetch_add(&bar[0], 1u, __ATOMIC_ACQ_REL,
                                             __HIP_MEMORY_SCOPE_AGENT);
      if (prev + 1u == target) {
        __hip_atomic_store(&bar[1], (unsigned)(t + 1), __ATOMIC_RELEASE,
                           __HIP_MEMORY_SCOPE_AGENT);
      } else {
        while (__hip_atomic_load(&bar[1], __ATOMIC_ACQUIRE,
                                 __HIP_MEMORY_SCOPE_AGENT) < (unsigned)(t + 1)) {
          __builtin_amdgcn_s_sleep(1);
        }
      }
    }
    __syncthreads();
    __threadfence();   // acquire: invalidate stale h lines before next staging
  }
}

// out[128][256] f32 = h_final(bf16) @ Wout^T(bf16) + bout. One WG, 8 waves.
__global__ void __launch_bounds__(256)
out_gemm(const bf16_t* __restrict__ h, const bf16_t* __restrict__ W,
         const float* __restrict__ bout, float* __restrict__ out) {
  const int wave = threadIdx.x >> 5;
  const int lane = threadIdx.x & 31;
  const int mbase = wave * 16;
  for (int nt = 0; nt < ODIM / 16; ++nt) {
    v8f acc = {0, 0, 0, 0, 0, 0, 0, 0};
    for (int k = 0; k < HIDD; k += 32) {
      v16bf av = load_a_tile(h, HIDD, mbase, k);
      v16bf bv = load_b_tile(W, HIDD, nt * 16, k);
      acc = wmma_bf16(av, bv, acc);
    }
    const int n = nt * 16 + (lane & 15);
    const int m0 = mbase + ((lane >> 4) << 3);
    const float bb = bout[n];
#pragma unroll
    for (int v = 0; v < 8; ++v) out[(m0 + v) * ODIM + n] = acc[v] + bb;
  }
}

// ---------------------------------------------------------------------------
extern "C" void kernel_launch(void* const* d_in, const int* in_sizes, int n_in,
                              void* d_out, int out_size, void* d_ws, size_t ws_size,
                              hipStream_t stream) {
  (void)in_sizes; (void)n_in; (void)out_size; (void)ws_size;
  const float* x    = (const float*)d_in[0];
  const float* Wi   = (const float*)d_in[1];
  const float* bi   = (const float*)d_in[2];
  const float* Wg   = (const float*)d_in[3];
  const float* bg   = (const float*)d_in[4];
  const float* Wf   = (const float*)d_in[5];
  const float* bfv  = (const float*)d_in[6];
  const float* Wo   = (const float*)d_in[7];
  const float* bo   = (const float*)d_in[8];
  const float* Wout = (const float*)d_in[9];
  const float* bout = (const float*)d_in[10];

  // Workspace layout (all 256B-aligned), total ~67.5 MiB
  char* ws = (char*)d_ws;
  size_t off = 0;
  unsigned* bar  = (unsigned*)(ws + off); off += 256;
  bf16_t* Wcomb  = (bf16_t*)(ws + off);   off += (size_t)4 * HIDD * KDIM * sizeof(bf16_t);
  float*  bcomb  = (float*) (ws + off);   off += (size_t)4 * HIDD * sizeof(float);
  bf16_t* hbuf   = (bf16_t*)(ws + off);   off += (size_t)2 * BATCH * HIDD * sizeof(bf16_t);
  bf16_t* Woutbf = (bf16_t*)(ws + off);   off += (size_t)ODIM * HIDD * sizeof(bf16_t);
  bf16_t* xbf    = (bf16_t*)(ws + off);   off += (size_t)SEQT * BATCH * DIN * sizeof(bf16_t);

  pack_params<<<1024, 256, 0, stream>>>(Wi, Wg, Wf, Wo, bi, bg, bfv, bo, Wout,
                                        Wcomb, bcomb, Woutbf, hbuf, bar);
  pack_x<<<2048, 256, 0, stream>>>(x, xbf);
  lstm_scan<<<NWG, 256, 0, stream>>>(xbf, Wcomb, bcomb, hbuf, bar);
  const bf16_t* hfin = hbuf + (size_t)(SEQT & 1) * BATCH * HIDD;  // final h buffer
  out_gemm<<<1, 256, 0, stream>>>(hfin, Woutbf, bout, (float*)d_out);
}
